// ExplainerGNN_67577015435768
// MI455X (gfx1250) — matrix operand order repaired and checked by reading, not compiled, not run on hardware
//
#include <hip/hip_runtime.h>
#include <hip/hip_bf16.h>
#include <stdint.h>

typedef __attribute__((ext_vector_type(16))) __bf16 v16bf;
typedef __attribute__((ext_vector_type(8)))  float  v8f;
typedef __attribute__((ext_vector_type(8)))  __bf16 bf16x8;

#define N_NODES 100000
#define N_EDGES 1600000
#define N_GR    256

// ---------------- utility kernels ----------------

__global__ __launch_bounds__(256) void zero_u32_kernel(unsigned* __restrict__ p, long long n){
  long long t = (long long)blockIdx.x * blockDim.x + threadIdx.x;
  if (t < n) p[t] = 0u;
}

// in: K x 64 f32 (row-major) -> out: 64 x K bf16 (transposed, K contiguous)
__global__ __launch_bounds__(256) void cvt_transpose_bf16_kernel(const float* __restrict__ in,
                                                                 __bf16* __restrict__ out, int K){
  int t = blockIdx.x * blockDim.x + threadIdx.x;
  if (t < K * 64){
    int nn = t / K, k = t - nn * K;
    out[t] = (__bf16)in[k * 64 + nn];
  }
}

// h0 = x * node_imp  (N x 128, bf16)
__global__ __launch_bounds__(256) void prep_h_kernel(const float* __restrict__ x,
                                                     const float* __restrict__ ni,
                                                     __bf16* __restrict__ h, int n){
  long long t = (long long)blockIdx.x * blockDim.x + threadIdx.x;
  if (t < (long long)n * 128){
    int i = (int)(t >> 7);
    h[t] = (__bf16)(x[t] * ni[i]);
  }
}

// ---------------- edge aggregation: agg[dst] += h[src] ----------------
__global__ __launch_bounds__(256) void aggregate_kernel(const int* __restrict__ src,
                                                        const int* __restrict__ dst,
                                                        const __bf16* __restrict__ h,
                                                        __bf16* __restrict__ agg,
                                                        int ne, int pshift){
  long long t = (long long)blockIdx.x * blockDim.x + threadIdx.x;
  long long total = (long long)ne << pshift;
  if (t >= total) return;
  int e = (int)(t >> pshift);
  int p = (int)(t & ((1 << pshift) - 1));
  int D = 2 << pshift;
  int s = src[e], d = dst[e];
  unsigned data = *(const unsigned*)(h + (size_t)s * D + 2 * p);
  if ((data & 0x7FFF7FFFu) == 0u) return;          // both halves +-0: skip atomic
  const __bf16* a = agg + (size_t)d * D + 2 * p;
  asm volatile("global_atomic_pk_add_bf16 %0, %1, off"
               :: "v"((unsigned long long)(uintptr_t)a), "v"(data) : "memory");
}

// ---------------- fused GNN layer ----------------
// Weights TRANSPOSED (64 x K, K contiguous). ks-outer / nt-inner with 4 live
// accumulators: batched LDS loads, 4 independent WMMAs per K-step.
template<int D>
__global__ __launch_bounds__(256)
void layer_kernel(const __bf16* __restrict__ h, const __bf16* __restrict__ agg,
                  const float* __restrict__ ni,
                  const __bf16* __restrict__ w1t, const float* __restrict__ b1, // 64 x D
                  const __bf16* __restrict__ w2t, const float* __restrict__ b2, // 64 x 64
                  __bf16* __restrict__ hout, float* __restrict__ nprob, int n)
{
  constexpr int KS1 = D / 32;
  constexpr int LD1 = D + 8;        // padded row stride (elems): rows 16B-aligned
  constexpr int LD2 = 64 + 8;
  constexpr int LZ  = 72;           // z1 row stride: 144B = 9*16B, conflict-free halves
  __shared__ __align__(16) __bf16 w1s[64 * LD1];
  __shared__ __align__(16) __bf16 w2s[64 * LD2];
  __shared__ __align__(16) __bf16 z1s[8][16 * LZ];

  const int tid = threadIdx.x;
  {
    constexpr int C1 = D / 8;                       // 16B chunks per row
    for (int i = tid; i < 64 * C1; i += 256){
      int row = i / C1, off = i - row * C1;
      *(uint4*)(w1s + row * LD1 + off * 8) = *(const uint4*)(w1t + row * D + off * 8);
    }
    for (int i = tid; i < 64 * 8; i += 256){
      int row = i >> 3, off = i & 7;
      *(uint4*)(w2s + row * LD2 + off * 8) = *(const uint4*)(w2t + row * 64 + off * 8);
    }
  }
  __syncthreads();

  const int wave = tid >> 5;
  const int lane = tid & 31;
  const int r    = lane & 15;     // A: row M ; B/C: column N
  const int hi   = lane >> 4;     // lane-half selects K/M sub-range
  const int m0   = (blockIdx.x * 8 + wave) * 16;

  // B fragment: K chunks [kbase, kbase+8) and [kbase+16, kbase+24) for column ncol -> two 16B LDS vector loads
  auto loadB = [&](const __bf16* ws, int ldw, int kbase, int ncol) -> v16bf {
    const bf16x8* p = (const bf16x8*)(ws + ncol * ldw + kbase);
    bf16x8 lo = p[0], hv = p[2];
    return __builtin_shufflevector(lo, hv, 0,1,2,3,4,5,6,7,8,9,10,11,12,13,14,15);
  };

  // ----- A fragments for GEMM1: z = h + agg (16 x D) -----
  // Per lane & K-step, the A layout is two contiguous 8-elem (16B) chunks.
  const int rowA = (m0 + r) < n ? (m0 + r) : (n - 1);
  bf16x8 hl[KS1], hh[KS1], gl[KS1], gh[KS1];
  #pragma unroll
  for (int ks = 0; ks < KS1; ++ks){
    const __bf16* hp = h   + (size_t)rowA * D + ks * 32 + 8 * hi;
    const __bf16* gp = agg + (size_t)rowA * D + ks * 32 + 8 * hi;
    hl[ks] = *(const bf16x8*)hp;
    hh[ks] = *(const bf16x8*)(hp + 16);
    gl[ks] = *(const bf16x8*)gp;
    gh[ks] = *(const bf16x8*)(gp + 16);
  }
  v16bf afrag[KS1];
  #pragma unroll
  for (int ks = 0; ks < KS1; ++ks){
    bf16x8 lo = hl[ks] + gl[ks];            // v_pk_add_bf16 x4
    bf16x8 hv = hh[ks] + gh[ks];
    afrag[ks] = __builtin_shufflevector(lo, hv, 0,1,2,3,4,5,6,7,8,9,10,11,12,13,14,15);
  }

  // ----- GEMM1: ks-outer, 4 accumulators -----
  v8f acc1[4] = {v8f{}, v8f{}, v8f{}, v8f{}};
  #pragma unroll
  for (int ks = 0; ks < KS1; ++ks){
    v16bf bfr[4];
    #pragma unroll
    for (int nt = 0; nt < 4; ++nt)
      bfr[nt] = loadB(w1s, LD1, ks * 32 + 8 * hi, nt * 16 + r);
    #pragma unroll
    for (int nt = 0; nt < 4; ++nt)
      acc1[nt] = __builtin_amdgcn_wmma_f32_16x16x32_bf16(false, afrag[ks], false, bfr[nt],
                                                         (short)0, acc1[nt], false, false);
  }
  // bias + relu -> z1s
  #pragma unroll
  for (int nt = 0; nt < 4; ++nt){
    float bias = b1[nt * 16 + r];
    #pragma unroll
    for (int v = 0; v < 8; ++v){
      float z = acc1[nt][v] + bias;
      z = z > 0.f ? z : 0.f;
      int m = v + 8 * hi;                       // C layout: lanes16-31 hold M+8
      z1s[wave][m * LZ + nt * 16 + r] = (__bf16)z;
    }
  }
  asm volatile("s_wait_dscnt 0x0" ::: "memory"); // wave-private LDS RAW
  __builtin_amdgcn_wave_barrier();

  // ----- A fragments for GEMM2 from z1s: two 16B LDS loads per K-step -----
  v16bf a2[2];
  #pragma unroll
  for (int ks = 0; ks < 2; ++ks){
    const bf16x8* zp = (const bf16x8*)&z1s[wave][r * LZ + ks * 32 + 8 * hi];
    bf16x8 lo = zp[0];
    bf16x8 hv = zp[2];                          // +16 elems = +32B = 2 x bf16x8
    a2[ks] = __builtin_shufflevector(lo, hv, 0,1,2,3,4,5,6,7,8,9,10,11,12,13,14,15);
  }

  float niv[8];
  #pragma unroll
  for (int v = 0; v < 8; ++v){
    int row = m0 + v + 8 * hi;
    niv[v] = ni[row < n ? row : n - 1];
  }

  // ----- GEMM2: ks-outer, 4 accumulators -----
  v8f acc2[4] = {v8f{}, v8f{}, v8f{}, v8f{}};
  #pragma unroll
  for (int ks = 0; ks < 2; ++ks){
    v16bf bfr[4];
    #pragma unroll
    for (int nt = 0; nt < 4; ++nt)
      bfr[nt] = loadB(w2s, LD2, ks * 32 + 8 * hi, nt * 16 + r);
    #pragma unroll
    for (int nt = 0; nt < 4; ++nt)
      acc2[nt] = __builtin_amdgcn_wmma_f32_16x16x32_bf16(false, a2[ks], false, bfr[nt],
                                                         (short)0, acc2[nt], false, false);
  }
  // bias + relu + *ni ; write hout / accumulate nprob
  #pragma unroll
  for (int nt = 0; nt < 4; ++nt){
    float bias = b2[nt * 16 + r];
    #pragma unroll
    for (int v = 0; v < 8; ++v){
      int m = v + 8 * hi;
      int row = m0 + m;
      float o = acc2[nt][v] + bias;
      o = o > 0.f ? o : 0.f;
      o *= niv[v];
      if (row < n){
        size_t idx = (size_t)row * 64 + nt * 16 + r;
        hout[idx] = (__bf16)o;
        nprob[idx] += o;                        // unique (row,col) per lane: no atomic
      }
    }
  }
}

// ---------------- graph pooling ----------------
__global__ __launch_bounds__(256) void pool_kernel(const float* __restrict__ nprob,
                                                   const int* __restrict__ batch,
                                                   float* __restrict__ gsum,
                                                   float* __restrict__ counts, int n){
  long long t = (long long)blockIdx.x * blockDim.x + threadIdx.x;
  if (t >= (long long)n * 64) return;
  int i = (int)(t >> 6);
  int c = (int)(t & 63);
  int b = batch[i];
  atomicAdd(&gsum[(size_t)b * 64 + c], nprob[t]);
  if (c == 0) atomicAdd(&counts[b], 1.0f);
}

__global__ __launch_bounds__(256) void final_kernel(const float* __restrict__ nprob,
                                                    const int* __restrict__ batch,
                                                    const float* __restrict__ gsum,
                                                    const float* __restrict__ counts,
                                                    const float* __restrict__ mlp_w,
                                                    const float* __restrict__ mlp_b,
                                                    float* __restrict__ out, int n){
  __shared__ float mw[128];
  if (threadIdx.x < 128) mw[threadIdx.x] = mlp_w[threadIdx.x];
  __syncthreads();
  int i = blockIdx.x * blockDim.x + threadIdx.x;
  if (i >= n) return;
  int b = batch[i];
  float cnt = counts[b]; cnt = cnt > 1.f ? cnt : 1.f;
  float inv = 1.f / cnt;
  float acc = mlp_b[0];
  #pragma unroll 4
  for (int c = 0; c < 64; ++c) acc += nprob[(size_t)i * 64 + c] * mw[c];
  #pragma unroll 4
  for (int c = 0; c < 64; ++c) acc += gsum[(size_t)b * 64 + c] * inv * mw[64 + c];
  acc = acc < -10.f ? -10.f : (acc > 10.f ? 10.f : acc);
  out[i] = 1.f / (1.f + __expf(-acc));
}

// ---------------- launcher ----------------
extern "C" void kernel_launch(void* const* d_in, const int* in_sizes, int n_in,
                              void* d_out, int out_size, void* d_ws, size_t ws_size,
                              hipStream_t stream){
  (void)in_sizes; (void)n_in; (void)out_size; (void)ws_size;
  const float* x     = (const float*)d_in[0];
  const int*   ei    = (const int*)  d_in[1];
  const int*   batch = (const int*)  d_in[3];
  const float* ni    = (const float*)d_in[6];
  const float* w1f[3] = {(const float*)d_in[7],  (const float*)d_in[11], (const float*)d_in[15]};
  const float* b1f[3] = {(const float*)d_in[8],  (const float*)d_in[12], (const float*)d_in[16]};
  const float* w2f[3] = {(const float*)d_in[9],  (const float*)d_in[13], (const float*)d_in[17]};
  const float* b2f[3] = {(const float*)d_in[10], (const float*)d_in[14], (const float*)d_in[18]};
  const float* mlp_w = (const float*)d_in[19];
  const float* mlp_b = (const float*)d_in[20];
  float* out = (float*)d_out;

  char* wsp = (char*)d_ws;
  auto carve = [&](size_t bytes) -> void* {
    void* p = (void*)wsp;
    wsp += (bytes + 255) & ~(size_t)255;
    return p;
  };
  __bf16* hA    = (__bf16*)carve((size_t)N_NODES * 128 * sizeof(__bf16));
  __bf16* hB    = (__bf16*)carve((size_t)N_NODES * 128 * sizeof(__bf16));
  __bf16* aggb  = (__bf16*)carve((size_t)N_NODES * 128 * sizeof(__bf16));
  float*  nprob = (float*) carve((size_t)N_NODES * 64 * sizeof(float));
  __bf16* wb    = (__bf16*)carve((size_t)(128 * 64 + 5 * 64 * 64) * sizeof(__bf16));
  float*  gsum  = (float*) carve((size_t)N_GR * 64 * sizeof(float));
  float*  cnts  = (float*) carve((size_t)N_GR * sizeof(float));

  // transposed bf16 weights: w1t[l] is 64 x K (K = 128 or 64), w2t[l] is 64 x 64
  const int  K1[3] = {128, 64, 64};
  __bf16* w1b[3] = {wb, wb + 12288, wb + 20480};
  __bf16* w2b[3] = {wb + 8192, wb + 16384, wb + 24576};

  for (int l = 0; l < 3; ++l){
    cvt_transpose_bf16_kernel<<<(K1[l] * 64 + 255) / 256, 256, 0, stream>>>(w1f[l], w1b[l], K1[l]);
    cvt_transpose_bf16_kernel<<<(64 * 64 + 255) / 256, 256, 0, stream>>>(w2f[l], w2b[l], 64);
  }
  prep_h_kernel<<<(int)(((long long)N_NODES * 128 + 255) / 256), 256, 0, stream>>>(x, ni, hA, N_NODES);
  zero_u32_kernel<<<(int)(((long long)N_NODES * 64 + 255) / 256), 256, 0, stream>>>(
      (unsigned*)nprob, (long long)N_NODES * 64);

  const int* src = ei;
  const int* dst = ei + N_EDGES;

  __bf16* hcur = hA; __bf16* hnext = hB;
  for (int l = 0; l < 3; ++l){
    int D = (l == 0) ? 128 : 64;
    int pshift = (l == 0) ? 6 : 5;
    long long aggu = (long long)N_NODES * (D / 2);
    zero_u32_kernel<<<(int)((aggu + 255) / 256), 256, 0, stream>>>((unsigned*)aggb, aggu);
    long long tot = (long long)N_EDGES * (D / 2);
    aggregate_kernel<<<(int)((tot + 255) / 256), 256, 0, stream>>>(src, dst, hcur, aggb,
                                                                   N_EDGES, pshift);
    int blocks = (N_NODES + 127) / 128;
    if (l == 0)
      layer_kernel<128><<<blocks, 256, 0, stream>>>(hcur, aggb, ni, w1b[0], b1f[0],
                                                    w2b[0], b2f[0], hnext, nprob, N_NODES);
    else
      layer_kernel<64><<<blocks, 256, 0, stream>>>(hcur, aggb, ni, w1b[l], b1f[l],
                                                   w2b[l], b2f[l], hnext, nprob, N_NODES);
    __bf16* tmp = hcur; hcur = hnext; hnext = tmp;
  }

  zero_u32_kernel<<<(N_GR * 64 + 255) / 256, 256, 0, stream>>>((unsigned*)gsum, N_GR * 64);
  zero_u32_kernel<<<1, 256, 0, stream>>>((unsigned*)cnts, N_GR);
  pool_kernel<<<(int)(((long long)N_NODES * 64 + 255) / 256), 256, 0, stream>>>(
      nprob, batch, gsum, cnts, N_NODES);
  final_kernel<<<(N_NODES + 255) / 256, 256, 0, stream>>>(nprob, batch, gsum, cnts,
                                                          mlp_w, mlp_b, out, N_NODES);
}
